// I2G_17952963297888
// MI455X (gfx1250) — matrix operand order repaired and checked by reading, not compiled, not run on hardware
//
#include <hip/hip_runtime.h>

typedef __attribute__((ext_vector_type(16))) _Float16 v16h;
typedef __attribute__((ext_vector_type(8)))  float    v8f;
typedef int vi4 __attribute__((vector_size(16)));

#define B_  4
#define N_  32768
#define D_  64
#define S_  1024
#define K_  32
#define M_  (B_*S_*K_)        /* 131072 = 2^17 */
#define RAD2 (0.2f*0.2f)
#define EPS_ 1e-5f

#if __has_builtin(__builtin_amdgcn_global_load_async_to_lds_b128)
#define HAVE_ASYNC_LDS 1
#else
#define HAVE_ASYNC_LDS 0
#endif

// ---------------------------------------------------------------------------
// 1) Farthest point sampling: one block per batch, 1024 threads (32 wave32s),
//    each thread owns 32 points (dist + coords in registers).
//    Per-step argmax: 5 shfl_xor rounds in-wave (no barrier) + 32-entry
//    cross-wave stage -> only 2 __syncthreads per step.
//    Tie-break: smallest index wins (matches jnp.argmax first-occurrence).
// ---------------------------------------------------------------------------
__global__ __launch_bounds__(1024) void fps_kernel(const float* __restrict__ xyz,
                                                   int* __restrict__ fps_idx) {
  int b = blockIdx.x;
  int tid = threadIdx.x;
  int lane = tid & 31, wid = tid >> 5;
  const float* xx = xyz + (size_t)b * 3 * N_;
  const float* xy = xx + N_;
  const float* xz = xx + 2 * N_;

  __shared__ float wV[32];
  __shared__ int   wI[32];
  __shared__ int   sFar;

  float px[32], py[32], pz[32], dd[32];
#pragma unroll
  for (int i = 0; i < 32; ++i) {
    int n = tid + i * 1024;
    px[i] = xx[n]; py[i] = xy[n]; pz[i] = xz[n];
    dd[i] = 1e10f;
  }

  int far = 0;
  for (int step = 0; step < S_; ++step) {
    if (tid == 0) fps_idx[b * S_ + step] = far;
    float cx = xx[far], cy = xy[far], cz = xz[far];
    float best = -1.0f; int bidx = 0;
#pragma unroll
    for (int i = 0; i < 32; ++i) {
      float dx = px[i] - cx, dy = py[i] - cy, dz = pz[i] - cz;
      float d = dx * dx + dy * dy + dz * dz;
      float nd = fminf(dd[i], d);
      dd[i] = nd;
      if (nd > best) { best = nd; bidx = tid + i * 1024; }
    }
#pragma unroll
    for (int off = 16; off > 0; off >>= 1) {
      float ov = __shfl_xor(best, off, 32);
      int   oi = __shfl_xor(bidx, off, 32);
      if (ov > best || (ov == best && oi < bidx)) { best = ov; bidx = oi; }
    }
    if (lane == 0) { wV[wid] = best; wI[wid] = bidx; }
    __syncthreads();
    if (wid == 0) {
      best = wV[lane]; bidx = wI[lane];
#pragma unroll
      for (int off = 16; off > 0; off >>= 1) {
        float ov = __shfl_xor(best, off, 32);
        int   oi = __shfl_xor(bidx, off, 32);
        if (ov > best || (ov == best && oi < bidx)) { best = ov; bidx = oi; }
      }
      if (lane == 0) sFar = bidx;
    }
    __syncthreads();
    far = sFar;
  }
}

// ---------------------------------------------------------------------------
// 2) Gather sampled centers; write new_xyz [B,3,S] directly into d_out.
// ---------------------------------------------------------------------------
__global__ void gather_centers(const float* __restrict__ xyz, const int* __restrict__ fpsidx,
                               float* __restrict__ centers, float* __restrict__ out_xyz) {
  int i = blockIdx.x * blockDim.x + threadIdx.x;
  if (i >= B_ * S_) return;
  int b = i >> 10, s = i & 1023;
  int n = fpsidx[i];
  const float* xb = xyz + (size_t)b * 3 * N_;
  float cx = xb[n], cy = xb[N_ + n], cz = xb[2 * N_ + n];
  centers[i * 3 + 0] = cx; centers[i * 3 + 1] = cy; centers[i * 3 + 2] = cz;
  out_xyz[(b * 3 + 0) * S_ + s] = cx;
  out_xyz[(b * 3 + 1) * S_ + s] = cy;
  out_xyz[(b * 3 + 2) * S_ + s] = cz;
}

// ---------------------------------------------------------------------------
// 3) Ball query: one wave32 per query; ordered first-K_ in-radius indices
//    via ballot + prefix popcount, padded with N-1. Early exit.
// ---------------------------------------------------------------------------
__global__ __launch_bounds__(256) void ball_query(const float* __restrict__ xyz,
                                                  const float* __restrict__ centers,
                                                  int* __restrict__ ballidx) {
  int wave = (int)((blockIdx.x * blockDim.x + threadIdx.x) >> 5);
  int lane = threadIdx.x & 31;
  int b = wave >> 10;
  const float* xb = xyz + (size_t)b * 3 * N_;
  float cx = centers[wave * 3 + 0];
  float cy = centers[wave * 3 + 1];
  float cz = centers[wave * 3 + 2];
  int* outq = ballidx + (size_t)wave * K_;
  int count = 0;
  for (int base = 0; base < N_; base += 32) {
    int n = base + lane;
    float dx = xb[n] - cx, dy = xb[N_ + n] - cy, dz = xb[2 * N_ + n] - cz;
    float d = dx * dx + dy * dy + dz * dz;
    bool pred = (d <= RAD2);
    unsigned mask = (unsigned)__ballot(pred);
    int pos = count + __popc(mask & ((1u << lane) - 1u));
    if (pred && pos < K_) outq[pos] = n;
    count += __popc(mask);
    if (count >= K_) break;
  }
  if (count < K_) {
    for (int p = count + lane; p < K_; p += 32) outq[p] = N_ - 1;
  }
}

// ---------------------------------------------------------------------------
// 4) Build features (3 xyz-diff + 64 point channels, zero-padded to 96) and
//    write directly in WMMA B-fragment order.
// ---------------------------------------------------------------------------
#define KB0 3
__global__ __launch_bounds__(256) void build_feat(const float* __restrict__ xyz,
                                                  const float* __restrict__ points,
                                                  const float* __restrict__ centers,
                                                  const int* __restrict__ ballidx,
                                                  _Float16* __restrict__ featp) {
  int p = blockIdx.x * blockDim.x + threadIdx.x;
  if (p >= M_) return;
  int q = p >> 5;        // b*S + s
  int b = q >> 10;
  int n = ballidx[p];
  const float* xb = xyz + (size_t)b * 3 * N_;
  float h[96];
  h[0] = xb[n]            - centers[q * 3 + 0];
  h[1] = xb[N_ + n]       - centers[q * 3 + 1];
  h[2] = xb[2 * N_ + n]   - centers[q * 3 + 2];
  const float* pb = points + (size_t)b * D_ * N_;
#pragma unroll
  for (int c = 0; c < D_; ++c) h[3 + c] = pb[(size_t)c * N_ + n];
#pragma unroll
  for (int c = 67; c < 96; ++c) h[c] = 0.0f;

  int ptile = p >> 4, pn = p & 15;
#pragma unroll
  for (int g = 0; g < 6; ++g) {
    v16h chunk;
#pragma unroll
    for (int e = 0; e < 16; ++e) chunk[e] = (_Float16)h[g * 16 + e];
    int kb = g >> 1, half = g & 1;
    size_t off = ((size_t)(ptile * KB0 + kb)) * 512 + (size_t)(pn + 16 * half) * 16;
    *(v16h*)(featp + off) = chunk;
  }
}

// ---------------------------------------------------------------------------
// 5) Pack weights [Cout,Kact] into WMMA A-fragment order (16-bit A 16x32).
// ---------------------------------------------------------------------------
__global__ void pack_weights(const float* __restrict__ w, _Float16* __restrict__ wp,
                             int Cout, int Kact, int KBn) {
  int i = blockIdx.x * blockDim.x + threadIdx.x;
  int total = (Cout / 16) * KBn * 32 * 16;
  if (i >= total) return;
  int e    = i & 15;
  int lane = (i >> 4) & 31;
  int kb   = (i >> 9) % KBn;
  int ct   = (i >> 9) / KBn;
  int m = lane & 15, half = lane >> 4;
  int row = ct * 16 + m;
  int k = kb * 32 + (e + (half << 4));
  float v = (k < Kact) ? w[(size_t)row * Kact + k] : 0.0f;
  wp[i] = (_Float16)v;
}

// ---------------------------------------------------------------------------
// 6) GEMM: y[Cout, M] = Wpack x Xpack using v_wmma_f32_16x16x32_f16.
//    The packed weight block (<=16KB) is staged ONCE per workgroup into LDS
//    with ASYNCcnt-tracked global_load_async_to_lds_b128 (when available),
//    synchronized with s_wait_asynccnt + barrier; A-fragments then come from
//    LDS, B-fragments stream from global into registers.
// ---------------------------------------------------------------------------
template <int CT, int KBn>
__global__ __launch_bounds__(256) void gemm_wmma(const _Float16* __restrict__ wp,
                                                 const _Float16* __restrict__ xp,
                                                 float* __restrict__ y) {
  __shared__ __align__(32) _Float16 sw[CT * KBn * 512];

#if HAVE_ASYNC_LDS
  // 16B per lane per issue; CT*KBn*64 chunks cover the whole weight block.
  for (int cch = threadIdx.x; cch < CT * KBn * 64; cch += 256) {
    vi4 __attribute__((address_space(1)))* gsrc =
        (vi4 __attribute__((address_space(1)))*)((const char*)wp + (size_t)cch * 16);
    vi4 __attribute__((address_space(3)))* ldst =
        (vi4 __attribute__((address_space(3)))*)((char*)sw + (size_t)cch * 16);
    __builtin_amdgcn_global_load_async_to_lds_b128(gsrc, ldst, 0, 0);
  }
#if __has_builtin(__builtin_amdgcn_s_wait_asynccnt)
  __builtin_amdgcn_s_wait_asynccnt(0);
#else
  asm volatile("s_wait_asynccnt 0" ::: "memory");
#endif
  __syncthreads();
#else
  for (int cch = threadIdx.x; cch < CT * KBn * 512; cch += 256) sw[cch] = wp[cch];
  __syncthreads();
#endif

  int wib  = threadIdx.x >> 5;
  int lane = threadIdx.x & 31;
  int pt   = blockIdx.x * 8 + wib;          // 8192 position tiles total
  int half = lane >> 4, pn = lane & 15;
  int p = pt * 16 + pn;

  v16h bfr[KBn];
#pragma unroll
  for (int kb = 0; kb < KBn; ++kb)
    bfr[kb] = *(const v16h*)(xp + ((size_t)(pt * KBn + kb)) * 512 + (size_t)lane * 16);

#pragma unroll
  for (int ct = 0; ct < CT; ++ct) {
    v8f acc = {};
#pragma unroll
    for (int kb = 0; kb < KBn; ++kb) {
      v16h a = *(const v16h*)(sw + ((size_t)(ct * KBn + kb)) * 512 + (size_t)lane * 16);
      acc = __builtin_amdgcn_wmma_f32_16x16x32_f16(false, a, false, bfr[kb],
                                                   (short)0, acc, false, false);
    }
#pragma unroll
    for (int v = 0; v < 8; ++v) {
      int c = ct * 16 + v + 8 * half;       // C/D layout: VGPR v -> M = v + 8*half
      y[(size_t)c * M_ + p] = acc[v];
    }
  }
}

// ---------------------------------------------------------------------------
// 7) BN stats: one block per channel, deterministic fixed-order tree reduce.
// ---------------------------------------------------------------------------
__global__ __launch_bounds__(256) void bn_stats(const float* __restrict__ y,
                                                float* __restrict__ stats) {
  int c = blockIdx.x;
  __shared__ float ss[256], sq[256];
  const float* yc = y + (size_t)c * M_;
  float s = 0.0f, q2 = 0.0f;
  for (int i = threadIdx.x; i < M_; i += 256) {
    if (i + 4096 < M_) __builtin_prefetch(&yc[i + 4096], 0, 1);
    float v = yc[i];
    s += v; q2 += v * v;
  }
  ss[threadIdx.x] = s; sq[threadIdx.x] = q2;
  __syncthreads();
  for (int off = 128; off > 0; off >>= 1) {
    if (threadIdx.x < off) {
      ss[threadIdx.x] += ss[threadIdx.x + off];
      sq[threadIdx.x] += sq[threadIdx.x + off];
    }
    __syncthreads();
  }
  if (threadIdx.x == 0) { stats[2 * c] = ss[0]; stats[2 * c + 1] = sq[0]; }
}

// ---------------------------------------------------------------------------
// 8) BN apply + ReLU + repack to f16 B-fragment layout (KB=2) for next GEMM.
// ---------------------------------------------------------------------------
__global__ __launch_bounds__(256) void bn_apply_pack(const float* __restrict__ y,
                                                     const float* __restrict__ stats,
                                                     const float* __restrict__ g,
                                                     const float* __restrict__ t,
                                                     _Float16* __restrict__ xp, int CG) {
  int i = blockIdx.x * blockDim.x + threadIdx.x;   // total = CG * M_
  if (i >= CG * M_) return;
  int cg = i >> 17;             // channel group of 16 (uniform per block)
  int p  = i & (M_ - 1);
  v16h chunk;
#pragma unroll
  for (int e = 0; e < 16; ++e) {
    int c = cg * 16 + e;
    float mu  = stats[2 * c] * (1.0f / M_);
    float var = stats[2 * c + 1] * (1.0f / M_) - mu * mu;
    float sc = rsqrtf(var + EPS_) * g[c];
    float v = (y[(size_t)c * M_ + p] - mu) * sc + t[c];
    chunk[e] = (_Float16)fmaxf(v, 0.0f);
  }
  int kb = cg >> 1, half = cg & 1;
  size_t off = ((size_t)((p >> 4) * 2 + kb)) * 512 + (size_t)((p & 15) + 16 * half) * 16;
  *(v16h*)(xp + off) = chunk;
}

// ---------------------------------------------------------------------------
// 9) Final: BN apply + ReLU + max over nsample -> d_out new_points [B,128,S].
// ---------------------------------------------------------------------------
__global__ __launch_bounds__(256) void bn_max_out(const float* __restrict__ y,
                                                  const float* __restrict__ stats,
                                                  const float* __restrict__ g,
                                                  const float* __restrict__ t,
                                                  float* __restrict__ out) {
  int i = blockIdx.x * blockDim.x + threadIdx.x;   // B*128*S
  if (i >= B_ * 128 * S_) return;
  int s = i & 1023;
  int c = (i >> 10) & 127;
  int b = i >> 17;
  float mu  = stats[2 * c] * (1.0f / M_);
  float var = stats[2 * c + 1] * (1.0f / M_) - mu * mu;
  float sc = rsqrtf(var + EPS_) * g[c];
  float sh = t[c] - mu * sc;
  size_t base = (size_t)c * M_ + (size_t)((b << 10) + s) * K_;
  float m = 0.0f;                 // ReLU outputs are >= 0
#pragma unroll 8
  for (int k = 0; k < K_; ++k) {
    float v = fmaxf(y[base + k] * sc + sh, 0.0f);
    m = fmaxf(m, v);
  }
  out[((size_t)(b * 128 + c) << 10) + s] = m;
}

// ---------------------------------------------------------------------------
extern "C" void kernel_launch(void* const* d_in, const int* in_sizes, int n_in,
                              void* d_out, int out_size, void* d_ws, size_t ws_size,
                              hipStream_t stream) {
  const float* xyz    = (const float*)d_in[0];
  const float* points = (const float*)d_in[1];
  const float* w0 = (const float*)d_in[2];
  const float* g0 = (const float*)d_in[4];
  const float* t0 = (const float*)d_in[5];
  const float* w1 = (const float*)d_in[6];
  const float* g1 = (const float*)d_in[8];
  const float* t1 = (const float*)d_in[9];
  const float* w2 = (const float*)d_in[10];
  const float* g2 = (const float*)d_in[12];
  const float* t2 = (const float*)d_in[13];
  float* out = (float*)d_out;

  char* ws = (char*)d_ws;
  int*      fpsidx  = (int*)     (ws + 0);          // 16 KB
  float*    centers = (float*)   (ws + 16384);      // 48 KB
  int*      ballidx = (int*)     (ws + 65536);      // 512 KB
  _Float16* wp0     = (_Float16*)(ws + 589824);     // 12 KB
  _Float16* wp1     = (_Float16*)(ws + 602112);     // 8 KB
  _Float16* wp2     = (_Float16*)(ws + 610304);     // 16 KB
  _Float16* featp   = (_Float16*)(ws + 655360);     // 24 MB (feat, then x1, then x2)
  float*    y       = (float*)   (ws + 26214400);   // 64 MB (128 x M f32)
  float*    stats   = (float*)   (ws + 93323264);   // 1 KB
  // total ~89 MB

  fps_kernel   <<<B_, 1024, 0, stream>>>(xyz, fpsidx);
  gather_centers<<<(B_ * S_ + 255) / 256, 256, 0, stream>>>(xyz, fpsidx, centers, out);
  ball_query   <<<(B_ * S_ * 32) / 256, 256, 0, stream>>>(xyz, centers, ballidx);
  build_feat   <<<M_ / 256, 256, 0, stream>>>(xyz, points, centers, ballidx, featp);

  pack_weights<<<(4 * 3 * 512 + 255) / 256, 256, 0, stream>>>(w0, wp0, 64, 67, 3);
  pack_weights<<<(4 * 2 * 512 + 255) / 256, 256, 0, stream>>>(w1, wp1, 64, 64, 2);
  pack_weights<<<(8 * 2 * 512 + 255) / 256, 256, 0, stream>>>(w2, wp2, 128, 64, 2);

  // Layer 0: 64 x 96(pad of 67) x M
  gemm_wmma<4, 3><<<(M_ / 16) / 8, 256, 0, stream>>>(wp0, featp, y);
  bn_stats<<<64, 256, 0, stream>>>(y, stats);
  bn_apply_pack<<<(4 * M_) / 256, 256, 0, stream>>>(y, stats, g0, t0, featp, 4);

  // Layer 1: 64 x 64 x M
  gemm_wmma<4, 2><<<(M_ / 16) / 8, 256, 0, stream>>>(wp1, featp, y);
  bn_stats<<<64, 256, 0, stream>>>(y, stats);
  bn_apply_pack<<<(4 * M_) / 256, 256, 0, stream>>>(y, stats, g1, t1, featp, 4);

  // Layer 2: 128 x 64 x M, then BN + ReLU + max over K -> output
  gemm_wmma<8, 2><<<(M_ / 16) / 8, 256, 0, stream>>>(wp2, featp, y);
  bn_stats<<<128, 256, 0, stream>>>(y, stats);
  bn_max_out<<<(B_ * 128 * S_) / 256, 256, 0, stream>>>(y, stats, g2, t2, out + B_ * 3 * S_);
}
